// LSTM_24859270709600
// MI455X (gfx1250) — matrix operand order repaired and checked by reading, not compiled, not run on hardware
//
#include <hip/hip_runtime.h>
#include <hip/hip_bf16.h>

// ---------------------------------------------------------------------------
// 2-layer LSTM (B=128, T=1024, D=128, H=256) + FC head, for MI455X (gfx1250).
// Persistent-RNN: 8 workgroups x 16 batch rows, 16 wave32 waves each.
// Weights stream from L2 every step (1.75 MB f16, L2-resident, shared by all
// WGs); laundered global-addrspace pointers -> global_load_b128, no LICM
// hoist/spill. LDS rows padded -> conflict-free ds_load_b128. Depth-2
// software pipeline enforced with __builtin_amdgcn_sched_group_barrier so the
// scheduler cannot collapse it to just-in-time loads. 2 barriers/step, half
// of layer-1's GEMM hoisted ahead of the mid-step barrier.
// ---------------------------------------------------------------------------

typedef __attribute__((ext_vector_type(16))) _Float16 v16h;
typedef __attribute__((ext_vector_type(8)))  _Float16 v8h;
typedef __attribute__((ext_vector_type(4)))  _Float16 v4h;
typedef __attribute__((ext_vector_type(8)))  float    v8f;
typedef __attribute__((ext_vector_type(4)))  float    v4f;

#define GSPACE __attribute__((address_space(1)))

#define LSTM_B  128
#define LSTM_T  1024
#define LSTM_D  128
#define LSTM_H  256
#define LSTM_G  1024   // 4*H
#define LDA_X   136    // 128 + 8 pad: 272 B row -> banks rotate by 4/row
#define LDA_H   264    // 256 + 8 pad: 528 B row -> banks rotate by 4/row

// sched_group_barrier masks
#define SGB_WMMA      0x008
#define SGB_VMEM_READ 0x020
#define SGB_DS_READ   0x100

__device__ __forceinline__ float sigm(float x) {
  return 1.0f / (1.0f + __expf(-x));
}
__device__ __forceinline__ float tanha(float x) {
  x = fminf(fmaxf(x, -15.0f), 15.0f);
  float e = __expf(2.0f * x);
  return (e - 1.0f) / (e + 1.0f);
}

// A fragment (16x32 f16): lane holds row m=lane%16; K packed as two 8-elem
// runs (ISA 16-bit A layout). Source: row-major LDS tile, row stride LDA.
__device__ __forceinline__ v16h load_a_frag(const _Float16* __restrict__ arow,
                                            int kb, int kh) {
  const _Float16* p = arow + kb + kh * 8;      // run 1: K = kb + 8*kh ..
  v8h lo = *(const v8h*)(p);
  v8h hi = *(const v8h*)(p + 16);              // run 2: K = kb + 16 + 8*kh ..
  v16h a;
#pragma unroll
  for (int i = 0; i < 8; ++i) { a[i] = lo[i]; a[i + 8] = hi[i]; }
  return a;
}

// B fragment (32x16 f16): lane holds column n=lane%16; K packed as two 8-elem
// runs matching A (K = kb + 8*kh .., kb + 16 + 8*kh ..). B[k][n] = W[n][k], so a
// row-major f16 weight row supplies two 16-byte runs (global/L2).
__device__ __forceinline__ v16h load_b_frag(GSPACE const _Float16* wrow,
                                            int kb, int kh) {
  GSPACE const _Float16* p = wrow + kb + kh * 8;
  v8h lo = *(GSPACE const v8h*)(p);
  v8h hi = *(GSPACE const v8h*)(p + 16);
  v16h b;
#pragma unroll
  for (int i = 0; i < 8; ++i) { b[i] = lo[i]; b[i + 8] = hi[i]; }
  return b;
}

// acc[g] += A(16xKW from LDS, row stride LDA) x W_g^T(KW x 16 slice, L2).
// Wave wv owns hidden slice j in [16*wv, 16*wv+16). Depth-2 software
// pipeline, enforced by sched_group_barrier: the fragment loads for k-step
// ks+1 are pinned between k-step ks's loads and its WMMAs, so each wave keeps
// a full fragment set (10 loads) in flight behind every WMMA group.
template <int KW, int LDA>
__device__ __forceinline__ void gemm_acc(v8f acc[4],
                                         const _Float16* __restrict__ a_base,
                                         const _Float16* wgt,
                                         int lane15, int kh, int wv) {
  // Launder as u64, rebuild as an explicit global-addrspace pointer:
  // (a) blocks LICM from hoisting the t-invariant weight loads (which would
  //     spill ~900 VGPRs to scratch), (b) keeps loads on the GLOBAL path
  //     (global_load_b128) instead of FLAT (which ticks LOADcnt AND DScnt).
  unsigned long long wa = (unsigned long long)wgt;
  asm volatile("" : "+s"(wa));
  GSPACE const _Float16* w1 = (GSPACE const _Float16*)wa;

  const _Float16* arow = a_base + lane15 * LDA;
  GSPACE const _Float16* brow[4];
#pragma unroll
  for (int g = 0; g < 4; ++g)
    brow[g] = w1 + (size_t)(g * LSTM_H + wv * 16 + lane15) * KW;

  constexpr int NK = KW / 32;
  v16h bf[2][4];
  v16h af[2];
#pragma unroll
  for (int g = 0; g < 4; ++g) bf[0][g] = load_b_frag(brow[g], 0, kh);
  af[0] = load_a_frag(arow, 0, kh);

#pragma unroll
  for (int ks = 0; ks < NK; ++ks) {
    const int cb = ks & 1, nb = cb ^ 1;
    if (ks + 1 < NK) {
#pragma unroll
      for (int g = 0; g < 4; ++g)
        bf[nb][g] = load_b_frag(brow[g], (ks + 1) * 32, kh);
      af[nb] = load_a_frag(arow, (ks + 1) * 32, kh);
    }
#pragma unroll
    for (int g = 0; g < 4; ++g) {
      acc[g] = __builtin_amdgcn_wmma_f32_16x16x32_f16(
          /*neg_a=*/false, af[cb], /*neg_b=*/false, bf[cb][g],
          /*c_mod=*/(short)0, acc[g], /*reuse_a=*/false, /*reuse_b=*/false);
    }
  }

  // ---- scheduling directive: enforce the depth-2 pipeline ----
  // prologue group: k-step 0 fragments
  __builtin_amdgcn_sched_group_barrier(SGB_VMEM_READ, 8, 0);
  __builtin_amdgcn_sched_group_barrier(SGB_DS_READ, 2, 0);
#pragma unroll
  for (int ks = 0; ks < NK; ++ks) {
    if (ks + 1 < NK) {   // k-step ks+1 fragments issue BEFORE ks's WMMAs
      __builtin_amdgcn_sched_group_barrier(SGB_VMEM_READ, 8, 0);
      __builtin_amdgcn_sched_group_barrier(SGB_DS_READ, 2, 0);
    }
    __builtin_amdgcn_sched_group_barrier(SGB_WMMA, 4, 0);
  }
}

// Register-resident LSTM cell: acc[0..3] = i,f,g,o pre-activations (minus
// bias) for (m = r + 8*kh, j). c stays in VGPRs; new h (f16) goes to the
// LDS back buffer.
__device__ __forceinline__ void lstm_pointwise(v8f acc[4], const float bv[4],
                                               float c[8],
                                               _Float16* __restrict__ hnxt,
                                               int j, int kh) {
#pragma unroll
  for (int r = 0; r < 8; ++r) {
    int m = r + 8 * kh;
    float ig = sigm(acc[0][r] + bv[0]);
    float fg = sigm(acc[1][r] + bv[1]);
    float gg = tanha(acc[2][r] + bv[2]);
    float og = sigm(acc[3][r] + bv[3]);
    float cv = fg * c[r] + ig * gg;
    c[r] = cv;
    hnxt[m * LDA_H + j] = (_Float16)(og * tanha(cv));
  }
}

__global__ __launch_bounds__(512, 1)
__attribute__((amdgpu_waves_per_eu(4)))
void lstm_scan_kernel(
    const float* __restrict__ x,
    const _Float16* __restrict__ wih0, const _Float16* __restrict__ whh0,
    const _Float16* __restrict__ wih1, const _Float16* __restrict__ whh1,
    const float* __restrict__ bias0, const float* __restrict__ bias1,
    const float* __restrict__ fcw, const float* __restrict__ fcb,
    float* __restrict__ out) {
  __shared__ _Float16 xb[2][16 * LDA_X];     // x slice, ping-pong   ~8.5 KB
  __shared__ _Float16 h0b[2][16 * LDA_H];    // layer-0 h, ping-pong ~16.5 KB
  __shared__ _Float16 h1b[2][16 * LDA_H];    // layer-1 h, ping-pong ~16.5 KB

  const int tid    = threadIdx.x;
  const int lane   = tid & 31;
  const int wv     = tid >> 5;       // wave 0..15 -> hidden slice
  const int lane15 = lane & 15;
  const int kh     = lane >> 4;      // half-wave selector
  const int bblk   = blockIdx.x * 16;
  const int j      = wv * 16 + lane15;

  // init h = 0 (both buffers, incl. pad), c = 0
  for (int i = tid; i < 16 * LDA_H; i += 512) {
    h0b[0][i] = (_Float16)0.0f; h0b[1][i] = (_Float16)0.0f;
    h1b[0][i] = (_Float16)0.0f; h1b[1][i] = (_Float16)0.0f;
  }
  float c0r[8], c1r[8];
#pragma unroll
  for (int r = 0; r < 8; ++r) { c0r[r] = 0.0f; c1r[r] = 0.0f; }

  // hoist combined bias (b_ih + b_hh) for this thread's 4 gate columns
  float b0v[4], b1v[4];
#pragma unroll
  for (int g = 0; g < 4; ++g) {
    b0v[g] = bias0[g * LSTM_H + j];
    b1v[g] = bias1[g * LSTM_H + j];
  }

  // per-thread x staging coordinates (512 threads x 4 floats = 16x128 slice)
  const int xoff = tid * 4;
  const int xm = xoff >> 7, xk = xoff & 127;
  const int xlds = xm * LDA_X + xk;
  const float* xcol = x + ((size_t)(bblk + xm) * LSTM_T) * LSTM_D + xk;

  // stage x_0 into front buffer
  {
    v4f xv = *(const v4f*)(xcol);
    v4h hv;
#pragma unroll
    for (int i = 0; i < 4; ++i) hv[i] = (_Float16)xv[i];
    *(v4h*)(&xb[0][xlds]) = hv;
  }
  __syncthreads();

  for (int t = 0; t < LSTM_T; ++t) {
    const int cur = t & 1, nxt = cur ^ 1;

    // ---- stage x_{t+1} into back buffer (hidden behind this step's GEMMs) --
    if (t + 1 < LSTM_T) {
      const float* xp = xcol + (size_t)(t + 1) * LSTM_D;
      v4f xv = *(const v4f*)xp;
      v4h hv;
#pragma unroll
      for (int i = 0; i < 4; ++i) hv[i] = (_Float16)xv[i];
      *(v4h*)(&xb[nxt][xlds]) = hv;
      if (t + 2 < LSTM_T) __builtin_prefetch(xp + LSTM_D, 0, 0);
    }

    // ---- layer 0: x_t @ Wih0^T + h0 @ Whh0^T + b -> h0[nxt] ----
    v8f acc[4];
#pragma unroll
    for (int g = 0; g < 4; ++g) acc[g] = 0.0f;   // first WMMA uses inline-0 C
    gemm_acc<LSTM_D, LDA_X>(acc, xb[cur],  wih0, lane15, kh, wv);
    gemm_acc<LSTM_H, LDA_H>(acc, h0b[cur], whh0, lane15, kh, wv);
    lstm_pointwise(acc, b0v, c0r, h0b[nxt], j, kh);

    // ---- layer 1 (recurrent half): h1[cur] @ Whh1^T -- does NOT depend on
    //      h0[nxt], so it runs before the barrier and overlaps other waves'
    //      layer-0 pointwise / barrier arrival.
    v8f acc1[4];
#pragma unroll
    for (int g = 0; g < 4; ++g) acc1[g] = 0.0f;
    gemm_acc<LSTM_H, LDA_H>(acc1, h1b[cur], whh1, lane15, kh, wv);
    __syncthreads();   // B1: h0[nxt] complete for all waves

    // ---- layer 1 (input half): + h0_t @ Wih1^T, then cell update ----
    gemm_acc<LSTM_H, LDA_H>(acc1, h0b[nxt], wih1, lane15, kh, wv);
    lstm_pointwise(acc1, b1v, c1r, h1b[nxt], j, kh);
    __syncthreads();   // B2: h1[nxt] complete, xb[nxt] staged
  }

  // ---- FC head: out[b] = h1_last[b,:] . fc_w + fc_b (O = 1) ----
  // T = 1024 (even): final h1 lives in buffer 0.
  if (tid < 16) {
    float s = fcb[0];
    for (int k = 0; k < LSTM_H; ++k)
      s += (float)h1b[0][tid * LDA_H + k] * fcw[k];
    out[bblk + tid] = s;
  }
}

// ---- prep kernels: fp32 -> f16 weights, combined biases ----
__global__ void k_cvt_f16(const float* __restrict__ s, _Float16* __restrict__ d,
                          int n) {
  int i = blockIdx.x * blockDim.x + threadIdx.x;
  if (i < n) d[i] = (_Float16)s[i];
}
__global__ void k_bias_add(const float* __restrict__ a,
                           const float* __restrict__ b, float* __restrict__ d,
                           int n) {
  int i = blockIdx.x * blockDim.x + threadIdx.x;
  if (i < n) d[i] = a[i] + b[i];
}

extern "C" void kernel_launch(void* const* d_in, const int* in_sizes, int n_in,
                              void* d_out, int out_size, void* d_ws,
                              size_t ws_size, hipStream_t stream) {
  const float* x     = (const float*)d_in[0];
  const float* w_ih0 = (const float*)d_in[1];
  const float* w_hh0 = (const float*)d_in[2];
  const float* b_ih0 = (const float*)d_in[3];
  const float* b_hh0 = (const float*)d_in[4];
  const float* w_ih1 = (const float*)d_in[5];
  const float* w_hh1 = (const float*)d_in[6];
  const float* b_ih1 = (const float*)d_in[7];
  const float* b_hh1 = (const float*)d_in[8];
  const float* fc_w  = (const float*)d_in[9];
  const float* fc_b  = (const float*)d_in[10];
  float* out = (float*)d_out;

  // workspace layout (f16 weights row-major + combined f32 biases): ~1.84 MB
  _Float16* wih0 = (_Float16*)d_ws;
  _Float16* whh0 = wih0 + LSTM_G * LSTM_D;
  _Float16* wih1 = whh0 + LSTM_G * LSTM_H;
  _Float16* whh1 = wih1 + LSTM_G * LSTM_H;
  float* bias0 = (float*)(whh1 + LSTM_G * LSTM_H);
  float* bias1 = bias0 + LSTM_G;

  const int nt = 256;
  auto blocks = [&](int n) { return (n + nt - 1) / nt; };
  k_cvt_f16<<<blocks(LSTM_G * LSTM_D), nt, 0, stream>>>(w_ih0, wih0,
                                                        LSTM_G * LSTM_D);
  k_cvt_f16<<<blocks(LSTM_G * LSTM_H), nt, 0, stream>>>(w_hh0, whh0,
                                                        LSTM_G * LSTM_H);
  k_cvt_f16<<<blocks(LSTM_G * LSTM_H), nt, 0, stream>>>(w_ih1, wih1,
                                                        LSTM_G * LSTM_H);
  k_cvt_f16<<<blocks(LSTM_G * LSTM_H), nt, 0, stream>>>(w_hh1, whh1,
                                                        LSTM_G * LSTM_H);
  k_bias_add<<<blocks(LSTM_G), nt, 0, stream>>>(b_ih0, b_hh0, bias0, LSTM_G);
  k_bias_add<<<blocks(LSTM_G), nt, 0, stream>>>(b_ih1, b_hh1, bias1, LSTM_G);

  // 8 persistent workgroups (one per 16 batch rows), 16 wave32 waves each
  lstm_scan_kernel<<<LSTM_B / 16, 512, 0, stream>>>(
      x, wih0, whh0, wih1, whh1, bias0, bias1, fc_w, fc_b, out);
}
